// RNN_44693429682561
// MI455X (gfx1250) — matrix operand order repaired and checked by reading, not compiled, not run on hardware
//
#include <hip/hip_runtime.h>
#include <hip/hip_bf16.h>

typedef __bf16 bf16;
typedef __attribute__((ext_vector_type(16))) __bf16 v16bf;
typedef __attribute__((ext_vector_type(8)))  __bf16 v8bf;
typedef __attribute__((ext_vector_type(8)))  float  v8f;

#define BB   4
#define SS   4096
#define DIN  2048
#define DD   2048
#define DOUT 2048
#define NH   32
#define HD   64

// ---------------------------------------------------------------------------
// helpers
// ---------------------------------------------------------------------------
static __device__ __forceinline__ v8f zero_v8f() {
    v8f z;
#pragma unroll
    for (int i = 0; i < 8; ++i) z[i] = 0.f;
    return z;
}

static __device__ __forceinline__ float silu_f(float x) {
    return x / (1.f + __expf(-x));
}

static __device__ __forceinline__ float block_reduce_sum256(float v) {
    __shared__ float red[256];
    red[threadIdx.x] = v;
    __syncthreads();
#pragma unroll
    for (int off = 128; off > 0; off >>= 1) {
        if ((int)threadIdx.x < off) red[threadIdx.x] += red[threadIdx.x + off];
        __syncthreads();
    }
    float r = red[0];
    __syncthreads();
    return r;
}

// ---------------------------------------------------------------------------
// one-pass f32 -> bf16 conversion (n must be a multiple of 2048)
// ---------------------------------------------------------------------------
__global__ __launch_bounds__(256) void cvt_f32_to_bf16(
        const float* __restrict__ src, bf16* __restrict__ dst, int n) {
    const int i = (blockIdx.x * 256 + threadIdx.x) * 8;
    if (i >= n) return;
    float4 f0 = *(const float4*)(src + i);
    float4 f1 = *(const float4*)(src + i + 4);
    v8bf o;
    o[0]=(__bf16)f0.x; o[1]=(__bf16)f0.y; o[2]=(__bf16)f0.z; o[3]=(__bf16)f0.w;
    o[4]=(__bf16)f1.x; o[5]=(__bf16)f1.y; o[6]=(__bf16)f1.z; o[7]=(__bf16)f1.w;
    *(v8bf*)(dst + i) = o;
}

// ---------------------------------------------------------------------------
// WMMA fragment loaders (all-bf16 path)
// A fragment 16x32 (MxK), row-major A, lda = K-stride.
// Lane layout (ISA 7.12.2): lanes 0-15: M=lane, elems 0-7 = K 0-7, elems 8-15 = K 16-23
//                           lanes 16-31: M=lane-16, elems 0-7 = K 8-15, 8-15 = K 24-31
// ---------------------------------------------------------------------------
static __device__ __forceinline__ v16bf load_a_bf16(const bf16* A, int lda, int mBase, int kb) {
    const int lane = threadIdx.x & 31;
    const int row  = mBase + (lane & 15);
    const int kOff = (lane & 16) ? 8 : 0;
    const bf16* p = A + (size_t)row * lda + kb + kOff;
    v8bf lo = *(const v8bf*)(p);        // K +0..7   (16B aligned)
    v8bf hi = *(const v8bf*)(p + 16);   // K +16..23
    __builtin_prefetch(p + 32, 0, 3);   // next K-step, same row (speculative)
    v16bf a;
#pragma unroll
    for (int i = 0; i < 8; ++i) { a[i] = lo[i]; a[8 + i] = hi[i]; }
    return a;
}

// B fragment 32x16 (KxN) with B = W^T, W row-major (N x K) in bf16:
// B[k][n] = W[n][k] -> per lane 16 K values (two runs of 8) of one W row.
// Lane layout: lanes 0-15: N=lane, K kb+0..7,16..23 ; lanes 16-31: N=lane-16, K kb+8..15,24..31.
static __device__ __forceinline__ v16bf load_b_bf16(const bf16* W, int ldw, int nBase, int kb) {
    const int lane = threadIdx.x & 31;
    const int col  = nBase + (lane & 15);
    const int kOff = (lane & 16) ? 8 : 0;
    const bf16* p = W + (size_t)col * ldw + kb + kOff;
    v8bf lo = *(const v8bf*)(p);        // K +0..7
    v8bf hi = *(const v8bf*)(p + 16);   // K +16..23
    __builtin_prefetch(p + 32, 0, 3);   // next K-step, same column
    v16bf b;
#pragma unroll
    for (int i = 0; i < 8; ++i) { b[i] = lo[i]; b[8 + i] = hi[i]; }
    return b;
}

// ---------------------------------------------------------------------------
// WMMA GEMM: Out = A(MxK) @ W(NxK)^T [+ bias], all operands bf16, f32 accum.
// block = 256 threads = 8 waves; block tile 128x128; wave tile 32x64.
// SPLIT_OUT: columns [0,splitN) -> out0 (bf16), rest -> out1 (bf16)
// else: out0 = float*, stride Nn.
// ---------------------------------------------------------------------------
template <bool SPLIT_OUT>
__global__ __launch_bounds__(256) void gemm_wmma(
        const bf16* __restrict__ Ap, const bf16* __restrict__ Wmat,
        const float* __restrict__ bias,
        void* __restrict__ out0, void* __restrict__ out1,
        int M, int Nn, int Kk, int splitN) {
    const int lane = threadIdx.x & 31;
    const int wid  = threadIdx.x >> 5;
    const int mBase = blockIdx.x * 128 + (wid >> 1) * 32;   // 4 waves along M
    const int nBase = blockIdx.y * 128 + (wid & 1) * 64;    // 2 waves along N

    v8f acc[2][4];
#pragma unroll
    for (int i = 0; i < 2; ++i)
#pragma unroll
        for (int j = 0; j < 4; ++j) acc[i][j] = zero_v8f();

    for (int kb = 0; kb < Kk; kb += 32) {
        v16bf af[2], bf[4];
#pragma unroll
        for (int i = 0; i < 2; ++i) af[i] = load_a_bf16(Ap, Kk, mBase + i * 16, kb);
#pragma unroll
        for (int j = 0; j < 4; ++j) bf[j] = load_b_bf16(Wmat, Kk, nBase + j * 16, kb);

#pragma unroll
        for (int i = 0; i < 2; ++i)
#pragma unroll
            for (int j = 0; j < 4; ++j)
                acc[i][j] = __builtin_amdgcn_wmma_f32_16x16x32_bf16(
                    false, af[i], false, bf[j], (short)0, acc[i][j], false, false);
    }

    // epilogue: C/D layout: VGPR r -> row M = r + (lane<16 ? 0 : 8), col N = lane&15
    const int rOff = (lane & 16) ? 8 : 0;
#pragma unroll
    for (int j = 0; j < 4; ++j) {
        const int col = nBase + j * 16 + (lane & 15);
        const float bv = bias ? bias[col] : 0.f;
#pragma unroll
        for (int i = 0; i < 2; ++i) {
#pragma unroll
            for (int r = 0; r < 8; ++r) {
                const int row = mBase + i * 16 + rOff + r;
                const float v = acc[i][j][r] + bv;
                if (SPLIT_OUT) {
                    if (col < splitN)
                        ((bf16*)out0)[(size_t)row * splitN + col] = (bf16)v;
                    else
                        ((bf16*)out1)[(size_t)row * splitN + (col - splitN)] = (bf16)v;
                } else {
                    ((float*)out0)[(size_t)row * Nn + col] = v;
                }
            }
        }
    }
}

// ---------------------------------------------------------------------------
// causal depthwise conv (K=4) + bias + SiLU + RMSNorm over D, per (b,s)
// ---------------------------------------------------------------------------
__global__ __launch_bounds__(256) void conv_norm(
        const bf16* __restrict__ u_raw, const float* __restrict__ conv_w,
        const float* __restrict__ conv_b, const float* __restrict__ inw,
        bf16* __restrict__ u_n) {
    const int bs = blockIdx.x;
    const int s  = bs & (SS - 1);
    const int tid = threadIdx.x;
    const size_t rowBase = (size_t)bs * DD;

    float v[8];
    float ss = 0.f;
#pragma unroll
    for (int i = 0; i < 8; ++i) {
        const int c = i * 256 + tid;
        float a = conv_b[c];
        const bf16* up = u_raw + rowBase + c;
#pragma unroll
        for (int k = 0; k < 4; ++k) {
            const int sp = s - 3 + k;
            if (sp >= 0) a += conv_w[c * 4 + k] * (float)up[(ptrdiff_t)(k - 3) * DD];
        }
        const float val = silu_f(a);
        v[i] = val;
        ss += val * val;
    }
    const float tot = block_reduce_sum256(ss);
    const float scale = rsqrtf(tot / (float)DD + 1e-6f);
#pragma unroll
    for (int i = 0; i < 8; ++i) {
        const int c = i * 256 + tid;
        u_n[rowBase + c] = (bf16)(v[i] * scale * inw[c]);
    }
}

// ---------------------------------------------------------------------------
// per-head matrix RNN: h_t = tanh(h_{t-1} @ Wn + u_t)
// one block per head n; 256 threads: b = t>>6, k = t&63.
// Wn column k in 64 VGPRs per lane; h state in LDS (double-buffered, float4
// reads -> ds_load_b128 broadcast; 4 accumulators for FMA ILP).
// ---------------------------------------------------------------------------
__global__ __launch_bounds__(256) void rnn_scan(
        const bf16* __restrict__ u_n, const float* __restrict__ state_w,
        bf16* __restrict__ states) {
    const int n = blockIdx.x;
    const int t = threadIdx.x;
    const int b = t >> 6;
    const int k = t & 63;

    __shared__ float colss[64];
    __shared__ __align__(16) float hbuf[2][BB][HD];

    // load column k of W_n (duplicated across the 4 b-groups)
    float wcol[HD];
    const float* Wp = state_w + (size_t)n * HD * HD;
#pragma unroll
    for (int j = 0; j < HD; ++j) wcol[j] = Wp[j * HD + k];

    // Frobenius norm of the head matrix
    if (b == 0) {
        float s = 0.f;
#pragma unroll
        for (int j = 0; j < HD; ++j) s += wcol[j] * wcol[j];
        colss[k] = s;
    }
    __syncthreads();
    float tot = 0.f;
#pragma unroll
    for (int j = 0; j < HD; ++j) tot += colss[j];
    const float inv = 1.f / sqrtf(tot);
#pragma unroll
    for (int j = 0; j < HD; ++j) wcol[j] *= inv;

    hbuf[0][b][k] = 0.f;
    __syncthreads();

    const size_t base = (size_t)b * SS * DD + (size_t)n * HD + k;
    int cur = 0;
    for (int step = 0; step < SS; ++step) {
        const size_t idx = base + (size_t)step * DD;
        float a0 = (float)u_n[idx], a1 = 0.f, a2 = 0.f, a3 = 0.f;
        const float4* hp4 = (const float4*)&hbuf[cur][b][0];
#pragma unroll
        for (int j = 0; j < HD / 4; ++j) {           // LDS b128 broadcast reads
            const float4 h4 = hp4[j];
            a0 += h4.x * wcol[4 * j + 0];
            a1 += h4.y * wcol[4 * j + 1];
            a2 += h4.z * wcol[4 * j + 2];
            a3 += h4.w * wcol[4 * j + 3];
        }
        const float hn = tanhf((a0 + a1) + (a2 + a3));
        states[idx] = (bf16)hn;
        hbuf[cur ^ 1][b][k] = hn;
        __syncthreads();
        cur ^= 1;
    }
}

// ---------------------------------------------------------------------------
// y = states * silu(gate); RMSNorm over D with norm_w; bf16 out for final GEMM
// ---------------------------------------------------------------------------
__global__ __launch_bounds__(256) void gate_norm(
        const bf16* __restrict__ states, const bf16* __restrict__ gate,
        const float* __restrict__ nw, bf16* __restrict__ y_n) {
    const int bs = blockIdx.x;
    const int tid = threadIdx.x;
    const size_t rowBase = (size_t)bs * DD;

    float v[8];
    float ss = 0.f;
#pragma unroll
    for (int i = 0; i < 8; ++i) {
        const int c = i * 256 + tid;
        const float g = (float)gate[rowBase + c];
        const float y = (float)states[rowBase + c] * silu_f(g);
        v[i] = y;
        ss += y * y;
    }
    const float tot = block_reduce_sum256(ss);
    const float scale = rsqrtf(tot / (float)DD + 1e-6f);
#pragma unroll
    for (int i = 0; i < 8; ++i) {
        const int c = i * 256 + tid;
        y_n[rowBase + c] = (bf16)(v[i] * scale * nw[c]);
    }
}

// ---------------------------------------------------------------------------
extern "C" void kernel_launch(void* const* d_in, const int* in_sizes, int n_in,
                              void* d_out, int out_size, void* d_ws, size_t ws_size,
                              hipStream_t stream) {
    const float* x       = (const float*)d_in[0];
    const float* w_in    = (const float*)d_in[1];
    const float* b_in    = (const float*)d_in[2];
    const float* conv_w  = (const float*)d_in[3];
    const float* conv_b  = (const float*)d_in[4];
    const float* inorm_w = (const float*)d_in[5];
    const float* state_w = (const float*)d_in[6];
    const float* norm_w  = (const float*)d_in[7];
    const float* w_out   = (const float*)d_in[8];
    float* out = (float*)d_out;

    const size_t nBSD = (size_t)BB * SS * DD;    // 33.5M elements
    // aliased slots (each nBSD bf16 = 67 MB):
    bf16* slot0 = (bf16*)d_ws;                   // x_bf16, then states
    bf16* slot1 = slot0 + nBSD;                  // u_raw, then y_n
    bf16* slot2 = slot1 + nBSD;                  // gate
    bf16* slot3 = slot2 + nBSD;                  // u_n
    bf16* w_in_bf  = slot3 + nBSD;               // (2D, Din) bf16  (16 MB)
    bf16* w_out_bf = w_in_bf + (size_t)(2 * DD) * DIN;  // (Dout, D) bf16 (8 MB)
    // total ~294 MB of d_ws

    const int M = BB * SS;  // 16384

    // 0) one-pass f32 -> bf16 of x and the two weight panels
    const int nx  = BB * SS * DIN;
    const int nw1 = 2 * DD * DIN;
    const int nw2 = DOUT * DD;
    cvt_f32_to_bf16<<<nx  / 2048, 256, 0, stream>>>(x,     slot0,    nx);
    cvt_f32_to_bf16<<<nw1 / 2048, 256, 0, stream>>>(w_in,  w_in_bf,  nw1);
    cvt_f32_to_bf16<<<nw2 / 2048, 256, 0, stream>>>(w_out, w_out_bf, nw2);

    // 1) input projection: h = x @ w_in^T + b_in  -> split into u_raw | gate
    dim3 g1(M / 128, (2 * DD) / 128);
    gemm_wmma<true><<<g1, 256, 0, stream>>>(slot0, w_in_bf, b_in, slot1, slot2,
                                            M, 2 * DD, DIN, DD);
    // 2) causal depthwise conv + SiLU + RMSNorm
    conv_norm<<<BB * SS, 256, 0, stream>>>(slot1, conv_w, conv_b, inorm_w, slot3);
    // 3) per-head matrix RNN scan (states -> slot0, x_bf16 now dead)
    rnn_scan<<<NH, 256, 0, stream>>>(slot3, state_w, slot0);
    // 4) gated output + RMSNorm (y_n -> slot1, u_raw now dead)
    gate_norm<<<BB * SS, 256, 0, stream>>>(slot0, slot2, norm_w, slot1);
    // 5) output projection: out = y_n @ w_out^T
    dim3 g2(M / 128, DOUT / 128);
    gemm_wmma<false><<<g2, 256, 0, stream>>>(slot1, w_out_bf, nullptr, out, nullptr,
                                             M, DOUT, DD, 0);
}
